// DestroyEdgewise_49787260895538
// MI455X (gfx1250) — compile-verified, matrix operand order checked by reading, not run on hardware
//
#include <hip/hip_runtime.h>

#define EMB 64
#define HID 32

typedef __attribute__((ext_vector_type(16))) __bf16 v16bf;
typedef __attribute__((ext_vector_type(8)))  float  v8f;

union PackBF { unsigned int u[8]; v16bf v; };

__device__ __forceinline__ float lrelu(float x) { return x > 0.f ? x : 0.01f * x; }

// round-to-nearest-even f32->bf16 for host-side-style packing in swizzle kernels
__device__ __forceinline__ unsigned short f2bf_bits(float f) {
  unsigned int u = __float_as_uint(f);
  u += 0x7FFFu + ((u >> 16) & 1u);
  return (unsigned short)(u >> 16);
}
__device__ __forceinline__ unsigned int pack2(float lo, float hi) {
  return (unsigned int)f2bf_bits(lo) | ((unsigned int)f2bf_bits(hi) << 16);
}

// A-operand (16x32 bf16, M x K) lane layout per CDNA5 ISA 7.12.2:
// lane L (h = L>=16, m = L&15):
//   VGPR 0..3 : K = colbase + 8*h + {0..7}        (bf16 pairs)
//   VGPR 4..7 : K = colbase + 16 + 8*h + {0..7}
// Native (__bf16) casts -> hardware cvt instructions instead of manual RNE.
__device__ __forceinline__ v16bf load_a_tile(const float* __restrict__ rp, int colbase, int h) {
  const float4 f0 = *(const float4*)(rp + colbase + 8 * h);
  const float4 f1 = *(const float4*)(rp + colbase + 8 * h + 4);
  const float4 f2 = *(const float4*)(rp + colbase + 16 + 8 * h);
  const float4 f3 = *(const float4*)(rp + colbase + 16 + 8 * h + 4);
  v16bf r;
  r[0]  = (__bf16)f0.x;  r[1]  = (__bf16)f0.y;  r[2]  = (__bf16)f0.z;  r[3]  = (__bf16)f0.w;
  r[4]  = (__bf16)f1.x;  r[5]  = (__bf16)f1.y;  r[6]  = (__bf16)f1.z;  r[7]  = (__bf16)f1.w;
  r[8]  = (__bf16)f2.x;  r[9]  = (__bf16)f2.y;  r[10] = (__bf16)f2.z;  r[11] = (__bf16)f2.w;
  r[12] = (__bf16)f3.x;  r[13] = (__bf16)f3.y;  r[14] = (__bf16)f3.z;  r[15] = (__bf16)f3.w;
  return r;
}

__device__ __forceinline__ v16bf load_w_tile(const unsigned int* __restrict__ p) {
  PackBF q;
  const uint4 a = *(const uint4*)(p);
  const uint4 b = *(const uint4*)(p + 4);
  q.u[0] = a.x; q.u[1] = a.y; q.u[2] = a.z; q.u[3] = a.w;
  q.u[4] = b.x; q.u[5] = b.y; q.u[6] = b.z; q.u[7] = b.w;
  return q.v;
}

#define WMMA_BF16(a, b, c) \
  __builtin_amdgcn_wmma_f32_16x16x32_bf16(false, (a), false, (b), (short)0, (c), false, false)

// ---------------------------------------------------------------------------
// Weight pre-swizzle: W[K][N] fp32 -> bf16 operand tiles [kt][nt][lane][8 dw].
// B operand uses the A lane layout applied to B^T (element (n,k)).
// ---------------------------------------------------------------------------
__global__ void swizzle_wgnn(const float* __restrict__ W, unsigned int* __restrict__ out,
                             int layers) {
  int t = blockIdx.x * 256 + threadIdx.x;
  int total = layers * 2 * 4 * 32 * 8;
  if (t >= total) return;
  int v = t & 7, lane = (t >> 3) & 31, nt = (t >> 8) & 3, kt = (t >> 10) & 1, l = t >> 11;
  int h = lane >> 4;
  int n = nt * 16 + (lane & 15);
  int kb = kt * 32 + ((v < 4) ? 0 : 16) + 8 * h + 2 * (v & 3);
  const float* Wl = W + (size_t)l * EMB * EMB;
  out[t] = pack2(Wl[kb * EMB + n], Wl[(kb + 1) * EMB + n]);
}

__global__ void swizzle_wedge(const float* __restrict__ Ws, const float* __restrict__ Wd,
                              unsigned int* __restrict__ out) {
  int t = blockIdx.x * 256 + threadIdx.x;   // total 4*4*32*8 = 4096
  if (t >= 4096) return;
  int v = t & 7, lane = (t >> 3) & 31, nt = (t >> 8) & 3, kt = (t >> 10) & 3;
  int h = lane >> 4;
  int n = nt * 16 + (lane & 15);
  int k = kt * 32 + ((v < 4) ? 0 : 16) + 8 * h + 2 * (v & 3);
  const float* W = (k < EMB) ? Ws : Wd;
  int kk = k & (EMB - 1);
  out[t] = pack2(W[kk * EMB + n], W[(kk + 1) * EMB + n]);
}

// ---------------------------------------------------------------------------
// h = coord @ Wn + bn   [N,2]x[2,64]
// ---------------------------------------------------------------------------
__global__ void node_embed(const float* __restrict__ coord, const float* __restrict__ Wn,
                           const float* __restrict__ bn, float* __restrict__ hbuf, int n) {
  int t = blockIdx.x * 256 + threadIdx.x;
  int node = t >> 6, c = t & 63;
  if (node >= n) return;
  float x = coord[2 * node], y = coord[2 * node + 1];
  hbuf[(size_t)node * EMB + c] = fmaf(x, Wn[c], fmaf(y, Wn[EMB + c], bn[c]));
}

// ---------------------------------------------------------------------------
// agg[dst] += h[src]  (segment_sum). 16 threads/edge, float4 per thread.
// h/agg are L2-resident (25.6 MB each << 192 MB L2).
// ---------------------------------------------------------------------------
__global__ void scatter_add(const float* __restrict__ hbuf, const int* __restrict__ esrc,
                            const int* __restrict__ edst, float* __restrict__ agg, int E) {
  long long t = (long long)blockIdx.x * 256 + threadIdx.x;
  int e = (int)(t >> 4);
  if (e >= E) return;
  int c = ((int)t & 15) * 4;
  const float* srow = hbuf + (size_t)esrc[e] * EMB + c;
  __builtin_prefetch(srow, 0, 1);             // global_prefetch_b8 into near caches
  const float4 v = *(const float4*)srow;
  float* a = agg + (size_t)edst[e] * EMB + c;
  atomicAdd(a + 0, v.x);
  atomicAdd(a + 1, v.y);
  atomicAdd(a + 2, v.z);
  atomicAdd(a + 3, v.w);
}

// ---------------------------------------------------------------------------
// h = h + lrelu(agg @ W + b).  One wave per 16-node tile; 8 WMMAs/tile.
// ---------------------------------------------------------------------------
__global__ void gnn_update(float* __restrict__ hbuf, const float* __restrict__ agg,
                           const unsigned int* __restrict__ wsw, const float* __restrict__ bias,
                           int n) {
  int tiles = (n + 15) >> 4;
  int wave = blockIdx.x * 8 + (threadIdx.x >> 5);
  if (wave >= tiles) return;                  // wave-uniform exit, EXEC stays all-ones
  int lane = threadIdx.x & 31;
  int h = lane >> 4, m = lane & 15;
  int base = wave * 16;
  int rrow = base + m; if (rrow >= n) rrow = n - 1;
  const float* arow = agg + (size_t)rrow * EMB;

  // warm the residual rows we will read after the WMMAs
  __builtin_prefetch(hbuf + (size_t)rrow * EMB, 0, 1);

  v16bf a0 = load_a_tile(arow, 0, h);
  v16bf a1 = load_a_tile(arow, 32, h);

  v8f acc[4];
#pragma unroll
  for (int nt = 0; nt < 4; ++nt) {
    v8f c = {};
    c = WMMA_BF16(a0, load_w_tile(wsw + ((0 * 4 + nt) * 32 + lane) * 8), c);
    c = WMMA_BF16(a1, load_w_tile(wsw + ((1 * 4 + nt) * 32 + lane) * 8), c);
    acc[nt] = c;
  }

  // D layout: vgpr r, lanes 0-15 -> M=r, lanes 16-31 -> M=8+r; N = lane&15
  int ncol = lane & 15;
#pragma unroll
  for (int nt = 0; nt < 4; ++nt) {
    int col = nt * 16 + ncol;
    float bv = bias[col];
#pragma unroll
    for (int r = 0; r < 8; ++r) {
      int node = base + r + 8 * h;
      if (node < n) {
        size_t idx = (size_t)node * EMB + col;
        hbuf[idx] = hbuf[idx] + lrelu(acc[nt][r] + bv);
      }
    }
  }
}

// ---------------------------------------------------------------------------
// Masked edge features only: ef[slot] = h[src]@We_src + h[dst]@We_dst + be
// == [h_src | h_dst] (16x128) @ Wcat (128x64).  16 WMMAs per 16-slot tile.
// ---------------------------------------------------------------------------
__global__ void edge_feat(const float* __restrict__ hbuf, const int* __restrict__ esrc,
                          const int* __restrict__ edst, const int* __restrict__ mask,
                          const unsigned int* __restrict__ wc, const float* __restrict__ be,
                          float* __restrict__ ef, int M) {
  int tiles = (M + 15) >> 4;
  int wave = blockIdx.x * 8 + (threadIdx.x >> 5);
  if (wave >= tiles) return;
  int lane = threadIdx.x & 31;
  int h = lane >> 4, m = lane & 15;
  int slot = wave * 16 + m; if (slot >= M) slot = M - 1;
  int e = mask[slot];
  const float* rs = hbuf + (size_t)esrc[e] * EMB;
  const float* rd = hbuf + (size_t)edst[e] * EMB;
  __builtin_prefetch(rs, 0, 1);
  __builtin_prefetch(rd, 0, 1);

  v16bf a[4];
  a[0] = load_a_tile(rs, 0, h);
  a[1] = load_a_tile(rs, 32, h);
  a[2] = load_a_tile(rd, 0, h);
  a[3] = load_a_tile(rd, 32, h);

  v8f acc[4];
#pragma unroll
  for (int nt = 0; nt < 4; ++nt) {
    v8f c = {};
#pragma unroll
    for (int kt = 0; kt < 4; ++kt)
      c = WMMA_BF16(a[kt], load_w_tile(wc + ((kt * 4 + nt) * 32 + lane) * 8), c);
    acc[nt] = c;
  }

  int ncol = lane & 15;
#pragma unroll
  for (int nt = 0; nt < 4; ++nt) {
    int col = nt * 16 + ncol;
    float bv = be[col];
#pragma unroll
    for (int r = 0; r < 8; ++r) {
      int sw = wave * 16 + r + 8 * h;
      if (sw < M) ef[(size_t)sw * EMB + col] = acc[nt][r] + bv;
    }
  }
}

// ---------------------------------------------------------------------------
// Per-sample MLP: y_j = lrelu(ef_j@W1+b1)@W2+b2 ; out = lrelu(y@Wd1+bd1)@Wd2+bd2
// One block (128 threads) per sample.
// ---------------------------------------------------------------------------
__global__ void mlp_final(const float* __restrict__ ef, const float* __restrict__ W1,
                          const float* __restrict__ b1, const float* __restrict__ W2,
                          const float* __restrict__ b2, const float* __restrict__ Wd1,
                          const float* __restrict__ bd1, const float* __restrict__ Wd2,
                          const float* __restrict__ bd2, float* __restrict__ out, int ES) {
  __shared__ float y[128];
  __shared__ float red[HID];
  int b = blockIdx.x, j = threadIdx.x;
  if (j < ES) {
    const float* row = ef + ((size_t)b * ES + j) * EMB;
    float xr[EMB];
#pragma unroll
    for (int k = 0; k < EMB; ++k) xr[k] = row[k];
    float acc = b2[0];
    for (int t = 0; t < HID; ++t) {
      float s = b1[t];
#pragma unroll
      for (int k = 0; k < EMB; ++k) s = fmaf(xr[k], W1[k * HID + t], s);
      acc = fmaf(lrelu(s), W2[t], acc);
    }
    y[j] = acc;
  }
  __syncthreads();
  if (j < HID) {
    float s = bd1[j];
    for (int e = 0; e < ES; ++e) s = fmaf(y[e], Wd1[e * HID + j], s);
    red[j] = lrelu(s) * Wd2[j];
  }
  __syncthreads();
  if (j == 0) {
    float s = bd2[0];
#pragma unroll
    for (int t = 0; t < HID; ++t) s += red[t];
    out[b] = s;
  }
}

// ---------------------------------------------------------------------------
extern "C" void kernel_launch(void* const* d_in, const int* in_sizes, int n_in,
                              void* d_out, int out_size, void* d_ws, size_t ws_size,
                              hipStream_t stream) {
  const float* coord  = (const float*)d_in[0];
  const int*   esrc   = (const int*)  d_in[1];
  const int*   edst   = (const int*)  d_in[2];
  const int*   mask   = (const int*)  d_in[3];
  const float* Wn     = (const float*)d_in[4];
  const float* bn     = (const float*)d_in[5];
  const float* Wgnn   = (const float*)d_in[6];
  const float* bgnn   = (const float*)d_in[7];
  const float* We_src = (const float*)d_in[8];
  const float* We_dst = (const float*)d_in[9];
  const float* be     = (const float*)d_in[10];
  const float* W1     = (const float*)d_in[11];
  const float* b1     = (const float*)d_in[12];
  const float* W2     = (const float*)d_in[13];
  const float* b2     = (const float*)d_in[14];
  const float* Wd1    = (const float*)d_in[15];
  const float* bd1    = (const float*)d_in[16];
  const float* Wd2    = (const float*)d_in[17];
  const float* bd2    = (const float*)d_in[18];
  float* out = (float*)d_out;

  const int N  = in_sizes[0] / 2;
  const int E  = in_sizes[1];
  const int M  = in_sizes[3];                 // B * EDGE_SIZE
  const int L  = in_sizes[6] / (EMB * EMB);   // gnn layers
  const int ES = in_sizes[15] / HID;          // EDGE_SIZE (94)
  const int B  = (ES > 0) ? M / ES : 0;

  // workspace carve (256B aligned)
  size_t off = 0;
  auto carve = [&](size_t bytes) { size_t o = off; off += (bytes + 255) & ~(size_t)255; return o; };
  char* ws = (char*)d_ws;
  float*        hbuf = (float*)(ws + carve((size_t)N * EMB * 4));
  float*        agg  = (float*)(ws + carve((size_t)N * EMB * 4));
  unsigned int* wg   = (unsigned int*)(ws + carve((size_t)L * 2048 * 4));
  unsigned int* wc   = (unsigned int*)(ws + carve((size_t)4096 * 4));
  float*        ef   = (float*)(ws + carve((size_t)M * EMB * 4));

  // 1) weight pre-swizzle into WMMA B-operand layout (bf16)
  {
    int tot = L * 2048;
    swizzle_wgnn<<<(tot + 255) / 256, 256, 0, stream>>>(Wgnn, wg, L);
    swizzle_wedge<<<16, 256, 0, stream>>>(We_src, We_dst, wc);
  }

  // 2) node embedding
  {
    long long tot = (long long)N * EMB;
    node_embed<<<(unsigned)((tot + 255) / 256), 256, 0, stream>>>(coord, Wn, bn, hbuf, N);
  }

  // 3) residual GNN layers: memset agg, scatter atomics, WMMA update
  const int tilesN = (N + 15) >> 4;
  for (int l = 0; l < L; ++l) {
    hipMemsetAsync(agg, 0, (size_t)N * EMB * 4, stream);
    long long st = (long long)E * 16;
    scatter_add<<<(unsigned)((st + 255) / 256), 256, 0, stream>>>(hbuf, esrc, edst, agg, E);
    gnn_update<<<(tilesN + 7) / 8, 256, 0, stream>>>(hbuf, agg, wg + (size_t)l * 2048,
                                                     bgnn + (size_t)l * EMB, N);
  }

  // 4) edge features for masked slots only (24k rows, not 1.6M)
  {
    int tilesM = (M + 15) >> 4;
    edge_feat<<<(tilesM + 7) / 8, 256, 0, stream>>>(hbuf, esrc, edst, mask, wc, be, ef, M);
  }

  // 5) per-sample MLP reduction -> out[B]
  if (B > 0)
    mlp_final<<<B, 128, 0, stream>>>(ef, W1, b1, W2, b2, Wd1, bd1, Wd2, bd2, out, ES);
}
